// SequentialTemporalBlock_81956565942812
// MI455X (gfx1250) — matrix operand order repaired and checked by reading, not compiled, not run on hardware
//
#include <hip/hip_runtime.h>
#include <cstdint>
#include <cstddef>

typedef __attribute__((ext_vector_type(16))) _Float16 v16h;
typedef __attribute__((ext_vector_type(8)))  float    v8f;

#define D_MODEL 512
#define HEADS   8
#define DH      64
#define D_FF    2048
#define BATCH   8
#define SEQ     2048
#define ROWS    (BATCH * SEQ)   // 16384

#if defined(__has_builtin)
#if __has_builtin(__builtin_amdgcn_tensor_load_to_lds) && __has_builtin(__builtin_amdgcn_s_wait_tensorcnt)
#define HAVE_TDM 1
#endif
#endif
#ifndef HAVE_TDM
#define HAVE_TDM 0
#endif

// ---------------------------------------------------------------------------
// WMMA fragment helpers (wave32; A and B share contiguous k = 16*(lane>>4) + s)
// ---------------------------------------------------------------------------

// A fragment: 16x32 f16 from row-major source (lda elements).
// lane (g=lane>>4, m=lane&15): VGPR v holds K = {base,base+1},
// base = 16g + 2v.  -> one contiguous 32B run per lane.
static __device__ __forceinline__ v16h load_frag_a(const _Float16* __restrict__ src, int lda) {
    const int l = threadIdx.x & 31;
    const int g = l >> 4, m = l & 15;
    const _Float16* row = src + (size_t)m * lda;
    v16h a;
#pragma unroll
    for (int v = 0; v < 8; ++v) {
        const int k = (g << 4) + (v << 1);
        a[2 * v]     = row[k];
        a[2 * v + 1] = row[k + 1];
    }
    return a;
}

// Pre-packed fragment: 16 contiguous halves per lane (32B, 16B aligned).
static __device__ __forceinline__ v16h load_frag_packed(const _Float16* p) {
    p = (const _Float16*)__builtin_assume_aligned(p, 16);
    v16h r;
#pragma unroll
    for (int i = 0; i < 16; ++i) r[i] = p[i];
    return r;
}

static __device__ __forceinline__ v8f wmma_f16(v16h a, v16h b, v8f c) {
    return __builtin_amdgcn_wmma_f32_16x16x32_f16(false, a, false, b, (short)0, c, false, false);
}

// ---------------------------------------------------------------------------
// Tensor Data Mover: 1-D async copy of n64*8 bytes global -> LDS.
// D# per CDNA5 ISA (count=1, type=2, data_size=8B).
// ---------------------------------------------------------------------------
#if HAVE_TDM
typedef __attribute__((ext_vector_type(4))) unsigned int u32x4;
typedef __attribute__((ext_vector_type(8))) int          i32x8;
typedef __attribute__((ext_vector_type(4))) int          i32x4;

static __device__ __forceinline__ unsigned rfl(unsigned x) {
    return (unsigned)__builtin_amdgcn_readfirstlane((int)x);
}

static __device__ __forceinline__ void tdm_load_1d(unsigned lds_off, const void* gptr, unsigned n64) {
    const uint64_t ga = (uint64_t)(uintptr_t)gptr;
    u32x4 g0;
    g0[0] = rfl(1u);                                   // count=1, user descriptor
    g0[1] = rfl(lds_off);                              // lds_addr (bytes)
    g0[2] = rfl((unsigned)ga);                         // global_addr[31:0]
    g0[3] = rfl(((unsigned)(ga >> 32) & 0x01FFFFFFu) | (2u << 30)); // ga[56:32] | type=2
    i32x8 g1;
    g1[0] = (int)rfl(3u << 16);                        // data_size = 8B
    g1[1] = (int)rfl((n64 & 0xFFFFu) << 16);           // tensor_dim0[15:0]  (bits 63:48)
    g1[2] = (int)rfl(n64 >> 16);                       // tensor_dim0[31:16] (bits 79:64)
    g1[3] = (int)rfl((n64 & 0xFFFFu) << 16);           // tile_dim0 (bits 127:112)
    g1[4] = 0;                                         // tile_dim1 = tile_dim2 = 0 (1-D)
    g1[5] = (int)rfl(n64);                             // tensor_dim0_stride[31:0]
    g1[6] = 0;
    g1[7] = 0;
    i32x4 z4 = {0, 0, 0, 0};
#if defined(__clang_major__) && (__clang_major__ >= 23)
    i32x8 z8 = {0, 0, 0, 0, 0, 0, 0, 0};
    __builtin_amdgcn_tensor_load_to_lds(g0, g1, z4, z4, z8, 0);
#else
    __builtin_amdgcn_tensor_load_to_lds(g0, g1, z4, z4, 0);
#endif
}
#endif  // HAVE_TDM

// ---------------------------------------------------------------------------
// Weight packer: f32 W[K][N] -> f16 fragment-packed
// layout: [wn = n/64][ks = k/32][t = 0..3][lane = 0..31][e = 0..15]
// element <-> (k = ks*32 + 2*(e>>1) + 16*(lane>>4) + (e&1), n = wn*64 + t*16 + (lane&15))
// ---------------------------------------------------------------------------
__global__ void pack_w_kernel(const float* __restrict__ W, _Float16* __restrict__ Wp,
                              int K, int N) {
    const int idx = blockIdx.x * blockDim.x + threadIdx.x;
    if (idx >= K * N) return;
    const int e    = idx & 15;
    const int lane = (idx >> 4) & 31;
    int q          = idx >> 9;
    const int t    = q & 3;  q >>= 2;
    const int ks   = q % (K / 32);
    const int wn   = q / (K / 32);
    const int k = ks * 32 + ((e >> 1) << 1) + ((lane >> 4) << 4) + (e & 1);
    const int c = wn * 64 + t * 16 + (lane & 15);
    Wp[idx] = (_Float16)W[(size_t)k * N + c];
}

// ---------------------------------------------------------------------------
// Wave-per-row LayerNorm, fp32 in -> fp16 out
// ---------------------------------------------------------------------------
__global__ __launch_bounds__(256) void ln_f16_kernel(const float* __restrict__ x,
                                                     const float* __restrict__ gamma,
                                                     const float* __restrict__ beta,
                                                     _Float16* __restrict__ out) {
    const int wave = (blockIdx.x * blockDim.x + threadIdx.x) >> 5;
    const int l = threadIdx.x & 31;
    const float* row = x + (size_t)wave * D_MODEL;
    float vals[16];
    float s = 0.f, s2 = 0.f;
#pragma unroll
    for (int i = 0; i < 16; ++i) {
        float v = row[l + 32 * i];
        vals[i] = v; s += v; s2 += v * v;
    }
#pragma unroll
    for (int o = 16; o >= 1; o >>= 1) {
        s  += __shfl_xor(s,  o, 32);
        s2 += __shfl_xor(s2, o, 32);
    }
    const float mu  = s * (1.f / D_MODEL);
    const float var = s2 * (1.f / D_MODEL) - mu * mu;
    const float r   = rsqrtf(var + 1e-5f);
    _Float16* orow = out + (size_t)wave * D_MODEL;
#pragma unroll
    for (int i = 0; i < 16; ++i) {
        const int c = l + 32 * i;
        orow[c] = (_Float16)((vals[i] - mu) * r * gamma[c] + beta[c]);
    }
}

// ---------------------------------------------------------------------------
// Generic WMMA GEMM, one wave = 16x64 tile, packed-B panel double-buffered
// through LDS by the Tensor Data Mover (fallback: direct packed global loads).
// MODE 0: outf = resid + (A W + bias)               (fp32 out)
// MODE 1: outh = gelu_exact(A W + bias)             (fp16 out)
// MODE 2: QKV epilogue: scatter +bias into Q row-major / K,V fragment-packed
// ---------------------------------------------------------------------------
template <int MODE, int NDIM, int KDIM>
__global__ __launch_bounds__(256) void gemm_kernel(const _Float16* __restrict__ A,
                                                   const _Float16* __restrict__ Wp,
                                                   const float* __restrict__ bias,
                                                   const float* __restrict__ resid,
                                                   float* __restrict__ outf,
                                                   _Float16* __restrict__ outh,
                                                   _Float16* __restrict__ qh,
                                                   _Float16* __restrict__ kp,
                                                   _Float16* __restrict__ vp) {
    const int TM  = ROWS / 16;
    const int gw  = (blockIdx.x * blockDim.x + threadIdx.x) >> 5;
    const int wm  = gw % TM;
    const int wn  = gw / TM;                 // uniform across the 8 waves of a block
    const int l   = threadIdx.x & 31;
    const int wid = threadIdx.x >> 5;
    (void)wid;

    const _Float16* Arow = A + (size_t)wm * 16 * KDIM;
    const _Float16* Bcol = Wp + (size_t)wn * KDIM * 64;   // (KDIM/32) blocks x 2048 halves
    constexpr int NSTEP = KDIM / 32;
    v8f acc[4] = {};

#if HAVE_TDM
    __shared__ __align__(16) _Float16 stage[8][2][2048];  // 4KB x 2 per wave
    const unsigned lds0 = (unsigned)(uintptr_t)(void*)&stage[wid][0][0];
    const unsigned lds1 = (unsigned)(uintptr_t)(void*)&stage[wid][1][0];
    tdm_load_1d(lds0, Bcol, 512);                         // 4096B = 512 x 8B
#pragma unroll 1
    for (int i = 0; i < NSTEP; ++i) {
        if (i + 1 < NSTEP) {
            tdm_load_1d(((i + 1) & 1) ? lds1 : lds0, Bcol + (size_t)(i + 1) * 2048, 512);
            __builtin_amdgcn_s_wait_tensorcnt((short)1);  // oldest (current buffer) done
        } else {
            __builtin_amdgcn_s_wait_tensorcnt((short)0);
        }
        asm volatile("" ::: "memory");
        v16h a = load_frag_a(Arow + i * 32, KDIM);
        const _Float16* sb = &stage[wid][i & 1][0];
#pragma unroll
        for (int t = 0; t < 4; ++t) {
            v16h b = load_frag_packed(sb + (t * 32 + l) * 16);
            acc[t] = wmma_f16(a, b, acc[t]);
        }
        asm volatile("" ::: "memory");
    }
#else
#pragma unroll 1
    for (int i = 0; i < NSTEP; ++i) {
        v16h a = load_frag_a(Arow + i * 32, KDIM);
        const _Float16* sb = Bcol + (size_t)i * 2048;
#pragma unroll
        for (int t = 0; t < 4; ++t) {
            v16h b = load_frag_packed(sb + (t * 32 + l) * 16);
            acc[t] = wmma_f16(a, b, acc[t]);
        }
    }
#endif

    const int g = l >> 4, n = l & 15;
#pragma unroll
    for (int t = 0; t < 4; ++t) {
        const int c = wn * 64 + t * 16 + n;
        const float bc = bias[c];
#pragma unroll
        for (int r = 0; r < 8; ++r) {
            const int rowi = wm * 16 + r + 8 * g;
            const float vv = acc[t][r] + bc;
            if (MODE == 0) {
                outf[(size_t)rowi * NDIM + c] = resid[(size_t)rowi * NDIM + c] + vv;
            } else if (MODE == 1) {
                const float ge = 0.5f * vv * (1.f + erff(vv * 0.70710678118f));
                outh[(size_t)rowi * NDIM + c] = (_Float16)ge;
            } else {
                // MODE 2: c in [0,1536): sel 0=Q 1=K 2=V
                const int sel = c / D_MODEL;
                const int cc  = c % D_MODEL;
                const int hh  = cc / DH, d = cc % DH;
                const int bb  = rowi >> 11;
                const int tok = rowi & (SEQ - 1);
                const size_t bh = (size_t)bb * HEADS + hh;
                const _Float16 hv = (_Float16)vv;
                if (sel == 0) {
                    qh[(bh * SEQ + tok) * DH + d] = hv;
                } else if (sel == 1) {
                    // K packed as B-fragments of K^T: [bh][jt=tok/16][kc=d/32][lane][16]
                    const int lane2 = (((d >> 4) & 1) << 4) | (tok & 15);
                    const int e2    = d & 15;
                    kp[(((bh * (SEQ / 16) + (tok >> 4)) * 2 + (d >> 5)) * 32 + lane2) * 16 + e2] = hv;
                } else {
                    // V packed as B-fragments: [bh][jb=tok/32][t2=d/16][lane][16]
                    const int lane2 = (((tok >> 4) & 1) << 4) | (d & 15);
                    const int e2    = tok & 15;
                    vp[(((bh * (SEQ / 32) + (tok >> 5)) * 4 + (d >> 4)) * 32 + lane2) * 16 + e2] = hv;
                }
            }
        }
    }
}

// ---------------------------------------------------------------------------
// Flash attention. One wave: 16 query rows x dh=64, online softmax over
// 32-key blocks. K/V arrive fragment-packed; P transits LDS in packed
// A-fragment layout. Row sums come from an extra WMMA against an all-ones
// B tile (P . 1 puts the row sum in every lane, same layout as O), so the
// only cross-lane reduction left is the running max.
// ---------------------------------------------------------------------------
__global__ __launch_bounds__(256) void attn_kernel(const _Float16* __restrict__ qh,
                                                   const _Float16* __restrict__ kp,
                                                   const _Float16* __restrict__ vp,
                                                   const uint8_t* __restrict__ mask,
                                                   _Float16* __restrict__ outh) {
    __shared__ __align__(16) _Float16 Pbuf[8][512];
    const int wid = threadIdx.x >> 5;
    const int gw  = blockIdx.x * 8 + wid;
    const int mt  = gw & (SEQ / 16 - 1);
    const int h   = (gw >> 7) & (HEADS - 1);
    const int b   = gw >> 10;
    const int l   = threadIdx.x & 31;
    const int g   = l >> 4, n = l & 15;

    const size_t bh = (size_t)b * HEADS + h;
    const _Float16* Q  = qh + (bh * SEQ + (size_t)mt * 16) * DH;
    const _Float16* Kb = kp + bh * (SEQ / 16) * 2 * 512;
    const _Float16* Vb = vp + bh * (SEQ / 32) * 4 * 512;
    const uint8_t* mrow_b = mask + (size_t)b * SEQ;

    const v16h aq0 = load_frag_a(Q, DH);
    const v16h aq1 = load_frag_a(Q + 32, DH);
    v16h bones;
#pragma unroll
    for (int i = 0; i < 16; ++i) bones[i] = (_Float16)1.0f;

    v8f o[4] = {};
    v8f o5   = {};                       // running row sums of exp (all lanes)
    float mrow[8];
#pragma unroll
    for (int r = 0; r < 8; ++r) mrow[r] = -3.0e38f;
    _Float16* P = Pbuf[wid];

#pragma unroll 1
    for (int j0 = 0; j0 < SEQ; j0 += 32) {
        const int jt = j0 >> 4;
        // ---- S = Q K^T (two 16x16 tiles, K-dim = dh) ----
        v8f s0 = {}, s1 = {};
        {
            v16h b00 = load_frag_packed(Kb + ((size_t)jt * 2 + 0) * 512 + l * 16);
            v16h b01 = load_frag_packed(Kb + ((size_t)jt * 2 + 1) * 512 + l * 16);
            s0 = wmma_f16(aq0, b00, s0);
            s0 = wmma_f16(aq1, b01, s0);
            v16h b10 = load_frag_packed(Kb + ((size_t)(jt + 1) * 2 + 0) * 512 + l * 16);
            v16h b11 = load_frag_packed(Kb + ((size_t)(jt + 1) * 2 + 1) * 512 + l * 16);
            s1 = wmma_f16(aq0, b10, s1);
            s1 = wmma_f16(aq1, b11, s1);
        }
        const bool k0m = mrow_b[j0 + n] != 0;
        const bool k1m = mrow_b[j0 + 16 + n] != 0;
        // destination lane/element for packed A-fragment P (cols n and 16+n)
        const int laneP = 0;
        const int eP    = n;
#pragma unroll
        for (int r = 0; r < 8; ++r) {
            float a0 = s0[r] * 0.125f; if (!k0m) a0 = -1e30f;
            float a1 = s1[r] * 0.125f; if (!k1m) a1 = -1e30f;
            float rv = fmaxf(a0, a1);
#pragma unroll
            for (int ofs = 8; ofs >= 1; ofs >>= 1) rv = fmaxf(rv, __shfl_xor(rv, ofs, 32));
            const float nm   = fmaxf(mrow[r], rv);
            const float p0   = __expf(a0 - nm);
            const float p1   = __expf(a1 - nm);
            const float corr = __expf(mrow[r] - nm);
            mrow[r] = nm;
#pragma unroll
            for (int t = 0; t < 4; ++t) o[t][r] *= corr;
            o5[r] *= corr;
            const int lr = laneP + r + 8 * g;          // dest lane in A-frag layout
            P[lr * 16 + eP]     = (_Float16)p0;        // col n     -> e = n
            P[(lr + 16) * 16 + eP] = (_Float16)p1;     // col 16+n  -> lane+16, e = n
        }
        asm volatile("s_wait_dscnt 0" ::: "memory");
        // ---- O += P V ; row sums += P . 1 ----
        const v16h ap = load_frag_packed(P + l * 16);
#pragma unroll
        for (int t = 0; t < 4; ++t) {
            v16h bv = load_frag_packed(Vb + (((size_t)(j0 >> 5)) * 4 + t) * 512 + l * 16);
            o[t] = wmma_f16(ap, bv, o[t]);
        }
        o5 = wmma_f16(ap, bones, o5);
        asm volatile("s_wait_dscnt 0" ::: "memory");
    }
#pragma unroll
    for (int r = 0; r < 8; ++r) {
        const float inv = 1.f / o5[r];
        const int tok = mt * 16 + r + 8 * g;
#pragma unroll
        for (int t = 0; t < 4; ++t) {
            outh[((size_t)b * SEQ + tok) * D_MODEL + h * DH + t * 16 + n] =
                (_Float16)(o[t][r] * inv);
        }
    }
}

// ---------------------------------------------------------------------------
// Host-side launch
// ---------------------------------------------------------------------------
extern "C" void kernel_launch(void* const* d_in, const int* in_sizes, int n_in,
                              void* d_out, int out_size, void* d_ws, size_t ws_size,
                              hipStream_t stream) {
    (void)in_sizes; (void)n_in; (void)out_size; (void)ws_size;
    const float*   x     = (const float*)d_in[0];
    const uint8_t* mask  = (const uint8_t*)d_in[1];
    const float*   ln1_g = (const float*)d_in[2];
    const float*   ln1_b = (const float*)d_in[3];
    const float*   w_qkv = (const float*)d_in[4];
    const float*   b_qkv = (const float*)d_in[5];
    const float*   w_out = (const float*)d_in[6];
    const float*   b_out = (const float*)d_in[7];
    const float*   ln2_g = (const float*)d_in[8];
    const float*   ln2_b = (const float*)d_in[9];
    const float*   w1    = (const float*)d_in[10];
    const float*   b1    = (const float*)d_in[11];
    const float*   w2    = (const float*)d_in[12];
    const float*   b2    = (const float*)d_in[13];
    float* out = (float*)d_out;

    char* ws = (char*)d_ws;
    size_t off = 0;
    auto alloc = [&](size_t bytes) -> void* {
        void* p = ws + off;
        off += (bytes + 255) & ~(size_t)255;
        return p;
    };
    _Float16* wqkv_p = (_Float16*)alloc((size_t)D_MODEL * 3 * D_MODEL * 2);
    _Float16* wout_p = (_Float16*)alloc((size_t)D_MODEL * D_MODEL * 2);
    _Float16* w1_p   = (_Float16*)alloc((size_t)D_MODEL * D_FF * 2);
    _Float16* w2_p   = (_Float16*)alloc((size_t)D_FF * D_MODEL * 2);
    _Float16* xh     = (_Float16*)alloc((size_t)ROWS * D_MODEL * 2);
    _Float16* qh     = (_Float16*)alloc((size_t)ROWS * D_MODEL * 2);
    _Float16* kp     = (_Float16*)alloc((size_t)ROWS * D_MODEL * 2);
    _Float16* vp     = (_Float16*)alloc((size_t)ROWS * D_MODEL * 2);
    _Float16* attn   = (_Float16*)alloc((size_t)ROWS * D_MODEL * 2);
    float*    y      = (float*)   alloc((size_t)ROWS * D_MODEL * 4);
    _Float16* s2h    = (_Float16*)alloc((size_t)ROWS * D_MODEL * 2);
    _Float16* hdn    = (_Float16*)alloc((size_t)ROWS * D_FF * 2);

    // 1. pack weights into WMMA B-fragment layout (f16)
    {
        int n;
        n = D_MODEL * 3 * D_MODEL;
        pack_w_kernel<<<(n + 255) / 256, 256, 0, stream>>>(w_qkv, wqkv_p, D_MODEL, 3 * D_MODEL);
        n = D_MODEL * D_MODEL;
        pack_w_kernel<<<(n + 255) / 256, 256, 0, stream>>>(w_out, wout_p, D_MODEL, D_MODEL);
        n = D_MODEL * D_FF;
        pack_w_kernel<<<(n + 255) / 256, 256, 0, stream>>>(w1, w1_p, D_MODEL, D_FF);
        n = D_FF * D_MODEL;
        pack_w_kernel<<<(n + 255) / 256, 256, 0, stream>>>(w2, w2_p, D_FF, D_MODEL);
    }
    // 2. LN1 -> f16
    ln_f16_kernel<<<ROWS / 8, 256, 0, stream>>>(x, ln1_g, ln1_b, xh);
    // 3. QKV projection, scattering into packed Q/K/V
    gemm_kernel<2, 3 * D_MODEL, D_MODEL>
        <<<(ROWS / 16) * (3 * D_MODEL / 64) / 8, 256, 0, stream>>>(
            xh, wqkv_p, b_qkv, nullptr, nullptr, nullptr, qh, kp, vp);
    // 4. flash attention
    attn_kernel<<<BATCH * HEADS * (SEQ / 16) / 8, 256, 0, stream>>>(qh, kp, vp, mask, attn);
    // 5. out-projection + residual -> y (fp32)
    gemm_kernel<0, D_MODEL, D_MODEL>
        <<<(ROWS / 16) * (D_MODEL / 64) / 8, 256, 0, stream>>>(
            attn, wout_p, b_out, x, y, nullptr, nullptr, nullptr, nullptr);
    // 6. LN2 -> f16
    ln_f16_kernel<<<ROWS / 8, 256, 0, stream>>>(y, ln2_g, ln2_b, s2h);
    // 7. FFN1 + exact GELU -> hdn (f16)
    gemm_kernel<1, D_FF, D_MODEL>
        <<<(ROWS / 16) * (D_FF / 64) / 8, 256, 0, stream>>>(
            s2h, w1_p, b1, nullptr, nullptr, hdn, nullptr, nullptr, nullptr);
    // 8. FFN2 + residual -> out (fp32)
    gemm_kernel<0, D_MODEL, D_FF>
        <<<(ROWS / 16) * (D_MODEL / 64) / 8, 256, 0, stream>>>(
            hdn, w2_p, b2, y, out, nullptr, nullptr, nullptr, nullptr);
}
